// CrossModalAttention_40630390620207
// MI455X (gfx1250) — matrix operand order repaired and checked by reading, not compiled, not run on hardware
//
#include <hip/hip_runtime.h>
#include <hip/hip_bf16.h>

// CrossModalAttention on gfx1250 (MI455X), wave32 + WMMA f32_16x16x32_f16.
// Pipeline: GEMM(q) -> GEMM(kv) -> flash-attention -> GEMM(out,+bias).
// Workspace layout (needs >= 64 MiB):
//   [0,16M)   q    f16 (8192 x 1024)
//   [16M,48M) kv   f16 (8192 x 2048)   k = cols [0,1024), v = cols [1024,2048)
//   [48M,64M) attn f16 (8192 x 1024)   (b,n, h*128+d)

typedef __attribute__((ext_vector_type(16))) _Float16 v16h;
typedef __attribute__((ext_vector_type(8)))  _Float16 v8h;
typedef __attribute__((ext_vector_type(2)))  _Float16 v2h;
typedef __attribute__((ext_vector_type(2)))  __fp16   v2fp;
typedef __attribute__((ext_vector_type(8)))  float    v8f;
typedef __attribute__((ext_vector_type(4)))  float    v4f;
typedef int vi4 __attribute__((vector_size(16)));

#define BATCH   4
#define SEQ     2048
#define IN_DIM  1024
#define DIM     1024
#define HEADS   8
#define DHEAD   128
#define MTOT    (BATCH * SEQ)      // 8192
#define SCALE_F 0.03125f           // DIM^-0.5 = 1/32

// ---------------- CDNA5 async global->LDS (probe-gated) ----------------
#if __has_builtin(__builtin_amdgcn_global_load_async_to_lds_b128) && \
    __has_builtin(__builtin_amdgcn_s_wait_asynccnt)
#define USE_ASYNC_LDS 1
#else
#define USE_ASYNC_LDS 0
#endif

// Copy 16 bytes global f16 -> LDS (async when available).
__device__ __forceinline__ void copy16B_g2s(const _Float16* g, _Float16* s) {
#if USE_ASYNC_LDS
  // param1 printed by clang as generic vi4*; AS-qualified pointers decay to
  // generic implicitly, so AS1/AS3 casts satisfy either signature variant.
  __builtin_amdgcn_global_load_async_to_lds_b128(
      (__attribute__((address_space(1))) vi4*)(void*)g,
      (__attribute__((address_space(3))) vi4*)(void*)s, 0, 0);
#else
  *(v8h*)s = *(const v8h*)g;
#endif
}
__device__ __forceinline__ void wait_async_copies() {
#if USE_ASYNC_LDS
  __builtin_amdgcn_s_wait_asynccnt(0);
#endif
}

// Pack 8 f32 -> 8 f16 with v_cvt_pk_rtz_f16_f32 when available.
__device__ __forceinline__ v8h pack8(v4f a, v4f b) {
#if __has_builtin(__builtin_amdgcn_cvt_pkrtz)
  v2h p0 = __builtin_bit_cast(v2h, __builtin_amdgcn_cvt_pkrtz(a[0], a[1]));
  v2h p1 = __builtin_bit_cast(v2h, __builtin_amdgcn_cvt_pkrtz(a[2], a[3]));
  v2h p2 = __builtin_bit_cast(v2h, __builtin_amdgcn_cvt_pkrtz(b[0], b[1]));
  v2h p3 = __builtin_bit_cast(v2h, __builtin_amdgcn_cvt_pkrtz(b[2], b[3]));
  v8h r;
  r[0] = p0[0]; r[1] = p0[1]; r[2] = p1[0]; r[3] = p1[1];
  r[4] = p2[0]; r[5] = p2[1]; r[6] = p3[0]; r[7] = p3[1];
  return r;
#else
  v8h r;
#pragma unroll
  for (int i = 0; i < 4; ++i) { r[i] = (_Float16)a[i]; r[i + 4] = (_Float16)b[i]; }
  return r;
#endif
}

// ---------------- WMMA helpers (ISA 16x16x32 f16 layouts) ----------------

__device__ __forceinline__ v8f wmma_f16(v16h a, v16h b, v8f c) {
  return __builtin_amdgcn_wmma_f32_16x16x32_f16(
      /*neg_a=*/false, a, /*neg_b=*/false, b,
      /*c_mod=*/(short)0, c, /*reuse_a=*/false, /*reuse_b=*/false);
}

// A-matrix 16x32 f16 fragment from row-major [m][k] buffer.
__device__ __forceinline__ v16h frag_a(const _Float16* base, int ldk, int m0, int k0) {
  const int lane = threadIdx.x & 31;
  const _Float16* p = base + (size_t)(m0 + (lane & 15)) * ldk + k0 + ((lane >> 4) << 3);
  v8h lo = *(const v8h*)(p);
  v8h hi = *(const v8h*)(p + 16);
  v16h a;
#pragma unroll
  for (int i = 0; i < 8; ++i) { a[i] = lo[i]; a[i + 8] = hi[i]; }
  return a;
}

// B-matrix 32x16 f16 fragment from row-major [n][k] buffer (B^T storage).
__device__ __forceinline__ v16h frag_b(const _Float16* base, int ldk, int n0, int k0) {
  const int lane = threadIdx.x & 31;
  const _Float16* p = base + (size_t)(n0 + (lane & 15)) * ldk + k0 + ((lane >> 4) << 3);
  v8h lo = *(const v8h*)(p);
  v8h hi = *(const v8h*)(p + 16);
  v16h b;
#pragma unroll
  for (int i = 0; i < 8; ++i) { b[i] = lo[i]; b[i + 8] = hi[i]; }
  return b;
}

// ---------------- Generic GEMM: C(MxN) = A(MxK) @ Bw(NxK)^T (+bias) ----------------
// Block tile 128(M) x 128(N), BK=32, 256 threads = 8 waves in 2(M) x 4(N);
// each wave computes 64x32 via 4x2 WMMA tiles (8 WMMA per k-step).

template <bool A_HALF, bool HAS_BIAS, bool OUT_HALF>
__global__ __launch_bounds__(256) void gemm_bt_wmma(
    const void* __restrict__ Av, const float* __restrict__ Bw,
    const float* __restrict__ bias, void* __restrict__ Cv,
    int M, int Nn, int K) {
  __shared__ alignas(16) _Float16 As[128 * 32];
  __shared__ alignas(16) _Float16 Bs[128 * 32];

  const int tid = threadIdx.x;
  const int lane = tid & 31;
  const int wave = tid >> 5;
  const int wm = wave >> 2;   // 0..1 -> 64 rows each
  const int wn = wave & 3;    // 0..3 -> 32 cols each
  const int blockM = blockIdx.y * 128;
  const int blockN = blockIdx.x * 128;

  const int row  = tid >> 1;        // 128 rows, 2 thr/row
  const int kcol = (tid & 1) * 16;  // 16 halfs each

  v8f acc[4][2] = {};

  for (int kk = 0; kk < K; kk += 32) {
    __syncthreads();
    // Stage A tile (128x32) -> f16 LDS
    if constexpr (A_HALF) {
      const _Float16* Ag = (const _Float16*)Av + (size_t)(blockM + row) * K + kk + kcol;
      copy16B_g2s(Ag,     &As[row * 32 + kcol]);
      copy16B_g2s(Ag + 8, &As[row * 32 + kcol + 8]);
      if (kk + 32 < K) __builtin_prefetch(Ag + 32, 0, 0);
    } else {
      const float* Ag = (const float*)Av + (size_t)(blockM + row) * K + kk + kcol;
      v4f a0 = *(const v4f*)Ag;
      v4f a1 = *(const v4f*)(Ag + 4);
      v4f a2 = *(const v4f*)(Ag + 8);
      v4f a3 = *(const v4f*)(Ag + 12);
      *(v8h*)&As[row * 32 + kcol]     = pack8(a0, a1);
      *(v8h*)&As[row * 32 + kcol + 8] = pack8(a2, a3);
      if (kk + 32 < K) __builtin_prefetch(Ag + 32, 0, 0);
    }
    // Stage B tile (128x32) weights f32 -> f16 LDS
    {
      const float* Bg = Bw + (size_t)(blockN + row) * K + kk + kcol;
      v4f b0 = *(const v4f*)Bg;
      v4f b1 = *(const v4f*)(Bg + 4);
      v4f b2 = *(const v4f*)(Bg + 8);
      v4f b3 = *(const v4f*)(Bg + 12);
      *(v8h*)&Bs[row * 32 + kcol]     = pack8(b0, b1);
      *(v8h*)&Bs[row * 32 + kcol + 8] = pack8(b2, b3);
      if (kk + 32 < K) __builtin_prefetch(Bg + 32, 0, 0);
    }
    wait_async_copies();
    __syncthreads();

    v16h af[4], bf[2];
#pragma unroll
    for (int mt = 0; mt < 4; ++mt) af[mt] = frag_a(As, 32, wm * 64 + mt * 16, 0);
#pragma unroll
    for (int nt = 0; nt < 2; ++nt) bf[nt] = frag_b(Bs, 32, wn * 32 + nt * 16, 0);
#pragma unroll
    for (int mt = 0; mt < 4; ++mt)
#pragma unroll
      for (int nt = 0; nt < 2; ++nt)
        acc[mt][nt] = wmma_f16(af[mt], bf[nt], acc[mt][nt]);
  }

  // Epilogue: C layout — VGPR r: lanes 0-15 (M=r, N=lane), lanes 16-31 (M=r+8).
#pragma unroll
  for (int mt = 0; mt < 4; ++mt)
#pragma unroll
    for (int nt = 0; nt < 2; ++nt)
#pragma unroll
      for (int r = 0; r < 8; ++r) {
        const int row_o = blockM + wm * 64 + mt * 16 + r + ((lane >> 4) << 3);
        const int col_o = blockN + wn * 32 + nt * 16 + (lane & 15);
        float v = acc[mt][nt][r];
        if constexpr (HAS_BIAS) v += bias[col_o];
        if constexpr (OUT_HALF)
          ((_Float16*)Cv)[(size_t)row_o * Nn + col_o] = (_Float16)v;
        else
          ((float*)Cv)[(size_t)row_o * Nn + col_o] = v;
      }
}

// ---------------- Flash attention ----------------
// grid = (B*HEADS, SEQ/128); 256 threads = 8 waves; wave owns 16 query rows.
// K tile 64x128 async-copied to LDS; V tile transposed to Vt[d][key] with
// packed b32 stores; P routed through per-wave LDS for C->A layout change.

__global__ __launch_bounds__(256) void attn_wmma(
    const _Float16* __restrict__ q, const _Float16* __restrict__ kv,
    _Float16* __restrict__ attn) {
  __shared__ alignas(16) _Float16 Ks[64 * 128];    // [key][d]
  __shared__ alignas(16) _Float16 Vt[128 * 64];    // [d][key]
  __shared__ alignas(16) _Float16 Ps[8 * 16 * 64]; // per-wave P tile [row][key]

  const int bh = blockIdx.x;
  const int b = bh >> 3;
  const int h = bh & 7;
  const int tid = threadIdx.x;
  const int lane = tid & 31;
  const int wave = tid >> 5;
  const int qrow0 = b * SEQ + blockIdx.y * 128 + wave * 16;  // absolute q row

  // Q fragments: 16 rows x 128 d, as 4 A-frags (K = d, 32 per frag)
  v16h qf[4];
#pragma unroll
  for (int kt = 0; kt < 4; ++kt)
    qf[kt] = frag_a(q, DIM, qrow0, h * DHEAD + kt * 32);

  float rmax[8], rsum[8], alpha[8];
  v8f o[8] = {};
#pragma unroll
  for (int r = 0; r < 8; ++r) { rmax[r] = -1e30f; rsum[r] = 0.0f; }

  // K-tile copy mapping: 4 thr/key-row, 32 halfs each.
  const int kr  = tid >> 2;
  const int kc0 = (tid & 3) * 32;
  // V-tile transpose mapping: thread -> key pair (kp,kp+1), 16 d values.
  const int kp = (tid & 31) * 2;
  const int dc = (tid >> 5) * 16;

  for (int j0 = 0; j0 < SEQ; j0 += 64) {
    __syncthreads();  // previous iteration done reading Ks/Vt
    {
      const _Float16* krow = kv + (size_t)(b * SEQ + j0 + kr) * (2 * DIM) + h * DHEAD + kc0;
#pragma unroll
      for (int i = 0; i < 4; ++i)
        copy16B_g2s(krow + i * 8, &Ks[kr * 128 + kc0 + i * 8]);
      const _Float16* v0 = kv + (size_t)(b * SEQ + j0 + kp) * (2 * DIM) + DIM + h * DHEAD + dc;
      const _Float16* v1 = v0 + 2 * DIM;
      v8h a0 = *(const v8h*)v0;
      v8h a1 = *(const v8h*)(v0 + 8);
      v8h b0 = *(const v8h*)v1;
      v8h b1 = *(const v8h*)(v1 + 8);
#pragma unroll
      for (int e = 0; e < 8; ++e) {
        v2h p0; p0[0] = a0[e]; p0[1] = b0[e];
        *(v2h*)&Vt[(dc + e) * 64 + kp] = p0;
        v2h p1; p1[0] = a1[e]; p1[1] = b1[e];
        *(v2h*)&Vt[(dc + 8 + e) * 64 + kp] = p1;
      }
      if (j0 + 64 < SEQ) {
        __builtin_prefetch(krow + (size_t)64 * 2 * DIM, 0, 0);
        __builtin_prefetch(v0 + (size_t)64 * 2 * DIM, 0, 0);
      }
    }
    wait_async_copies();
    __syncthreads();

    // S(16x64) = Q @ K^T, scaled
    v8f s[4] = {};
#pragma unroll
    for (int nt = 0; nt < 4; ++nt)
#pragma unroll
      for (int kt = 0; kt < 4; ++kt)
        s[nt] = wmma_f16(qf[kt], frag_b(Ks, 128, nt * 16, kt * 32), s[nt]);
#pragma unroll
    for (int nt = 0; nt < 4; ++nt)
#pragma unroll
      for (int r = 0; r < 8; ++r) s[nt][r] *= SCALE_F;

    // online softmax: row max over 4 n-tiles + 16 lanes of this half
#pragma unroll
    for (int r = 0; r < 8; ++r) {
      float mx = fmaxf(fmaxf(s[0][r], s[1][r]), fmaxf(s[2][r], s[3][r]));
#pragma unroll
      for (int off = 8; off >= 1; off >>= 1)
        mx = fmaxf(mx, __shfl_xor(mx, off, 32));
      const float mn = fmaxf(rmax[r], mx);
      alpha[r] = __expf(rmax[r] - mn);
      rmax[r] = mn;
    }
#pragma unroll
    for (int nt = 0; nt < 4; ++nt)
#pragma unroll
      for (int r = 0; r < 8; ++r) s[nt][r] = __expf(s[nt][r] - rmax[r]);
#pragma unroll
    for (int r = 0; r < 8; ++r) {
      float ls = s[0][r] + s[1][r] + s[2][r] + s[3][r];
#pragma unroll
      for (int off = 8; off >= 1; off >>= 1) ls += __shfl_xor(ls, off, 32);
      rsum[r] = rsum[r] * alpha[r] + ls;
    }
#pragma unroll
    for (int dt = 0; dt < 8; ++dt)
#pragma unroll
      for (int r = 0; r < 8; ++r) o[dt][r] *= alpha[r];

    // C-layout P -> LDS (per-wave region) -> A-layout frags
    _Float16* pw = Ps + wave * 16 * 64;
#pragma unroll
    for (int nt = 0; nt < 4; ++nt)
#pragma unroll
      for (int r = 0; r < 8; ++r) {
        const int prow = r + ((lane >> 4) << 3);
        const int pcol = nt * 16 + (lane & 15);
        pw[prow * 64 + pcol] = (_Float16)s[nt][r];
      }
    const v16h pf0 = frag_a(pw, 64, 0, 0);
    const v16h pf1 = frag_a(pw, 64, 0, 32);

    // O += P @ V   (B frag: N = d from Vt[d][key], K = key)
#pragma unroll
    for (int dt = 0; dt < 8; ++dt) {
      o[dt] = wmma_f16(pf0, frag_b(Vt, 64, dt * 16, 0),  o[dt]);
      o[dt] = wmma_f16(pf1, frag_b(Vt, 64, dt * 16, 32), o[dt]);
    }
  }

  // finalize and store f16 at (row, h*128 + d)
#pragma unroll
  for (int r = 0; r < 8; ++r) rsum[r] = 1.0f / rsum[r];
#pragma unroll
  for (int dt = 0; dt < 8; ++dt)
#pragma unroll
    for (int r = 0; r < 8; ++r) {
      const int row_o = qrow0 + r + ((lane >> 4) << 3);
      const int col_o = h * DHEAD + dt * 16 + (lane & 15);
      attn[(size_t)row_o * DIM + col_o] = (_Float16)(o[dt][r] * rsum[r]);
    }
}

// ---------------- Host launcher ----------------

extern "C" void kernel_launch(void* const* d_in, const int* in_sizes, int n_in,
                              void* d_out, int out_size, void* d_ws, size_t ws_size,
                              hipStream_t stream) {
  (void)in_sizes; (void)n_in; (void)out_size; (void)ws_size;
  const float* x_a   = (const float*)d_in[0];
  const float* x_b   = (const float*)d_in[1];
  const float* W_q   = (const float*)d_in[2];
  const float* W_kv  = (const float*)d_in[3];
  const float* W_out = (const float*)d_in[4];
  const float* b_out = (const float*)d_in[5];
  float* out = (float*)d_out;

  _Float16* q    = (_Float16*)d_ws;
  _Float16* kv   = (_Float16*)((char*)d_ws + (size_t)16 * 1024 * 1024);
  _Float16* attn = (_Float16*)((char*)d_ws + (size_t)48 * 1024 * 1024);

  dim3 blk(256);
  // q = x_a @ W_q^T : M=8192, N=1024, K=1024
  gemm_bt_wmma<false, false, true><<<dim3(DIM / 128, MTOT / 128), blk, 0, stream>>>(
      (const void*)x_a, W_q, nullptr, (void*)q, MTOT, DIM, IN_DIM);
  // kv = x_b @ W_kv^T : M=8192, N=2048, K=1024
  gemm_bt_wmma<false, false, true><<<dim3((2 * DIM) / 128, MTOT / 128), blk, 0, stream>>>(
      (const void*)x_b, W_kv, nullptr, (void*)kv, MTOT, 2 * DIM, IN_DIM);
  // attention
  attn_wmma<<<dim3(BATCH * HEADS, SEQ / 128), blk, 0, stream>>>(q, kv, attn);
  // out = attn @ W_out^T + b_out : M=8192, N=1024, K=1024
  gemm_bt_wmma<true, true, false><<<dim3(DIM / 128, MTOT / 128), blk, 0, stream>>>(
      (const void*)attn, W_out, b_out, (void*)out, MTOT, DIM, DIM);
}
